// GAT_ours_node_45492293599840
// MI455X (gfx1250) — compile-verified, matrix operand order checked by reading, not run on hardware
//

#include <hip/hip_runtime.h>
#include <hip/hip_bf16.h>

// ---- problem constants ----
#define BB   64
#define TT   128
#define BT   (BB*TT)      // 8192
#define EE   300
#define EP   320          // E padded to multiple of 32 for K-loop
#define HH   256
#define H4   1024         // 4*H
#define GG   512
#define FHH  256
#define NCC  2

typedef __attribute__((ext_vector_type(16))) _Float16 v16h;
typedef __attribute__((ext_vector_type(8)))  float    v8f;

__device__ __forceinline__ float sigf(float x) { return 1.0f / (1.0f + expf(-x)); }

#define WMMA_F16(A, B, C) \
    __builtin_amdgcn_wmma_f32_16x16x32_f16(false, (A), false, (B), (short)0, (C), false, false)

// ---------------------------------------------------------------------------
// Prep kernels
// ---------------------------------------------------------------------------

__global__ void k_embed_f16(const int* __restrict__ sentence,
                            const float* __restrict__ emb,
                            _Float16* __restrict__ X) {
    long idx = (long)blockIdx.x * blockDim.x + threadIdx.x;
    if (idx >= (long)BT * EP) return;
    int r = (int)(idx / EP);
    int e = (int)(idx % EP);
    X[idx] = (e < EE) ? (_Float16)emb[(long)sentence[r] * EE + e] : (_Float16)0.0f;
}

__global__ void k_cvt_pad_f16(const float* __restrict__ src,
                              _Float16* __restrict__ dst,
                              int rows, int scols, int dcols) {
    long idx = (long)blockIdx.x * blockDim.x + threadIdx.x;
    if (idx >= (long)rows * dcols) return;
    int r = (int)(idx / dcols);
    int c = (int)(idx % dcols);
    dst[idx] = (c < scols) ? (_Float16)src[(long)r * scols + c] : (_Float16)0.0f;
}

__global__ void k_bias_sum(const float* bihf, const float* bhhf,
                           const float* bihb, const float* bhhb,
                           float* __restrict__ bsum) {
    int i = blockIdx.x * blockDim.x + threadIdx.x;
    if (i < H4)            bsum[i] = bihf[i] + bhhf[i];
    else if (i < 2 * H4)   bsum[i] = bihb[i - H4] + bhhb[i - H4];
}

// ---------------------------------------------------------------------------
// Parallel input-transform GEMM: Gin[dir](t,b,:) = X(b*T+t,:) @ Wih[dir]^T + bsum
// Register-blocked: each wave computes a 2x4 block of 16x16 tiles (32x64 out).
// Per k-step: 2 A-frag + 4 B-frag loads feed 8 WMMAs. K = EP = 10 x 32.
// Output layout (T, B, 4H) f32 so the recurrence reads contiguous slabs.
// ---------------------------------------------------------------------------
__global__ void k_input_gemm(const _Float16* __restrict__ X,
                             const _Float16* __restrict__ Wih_all,
                             const float* __restrict__ bsum_all,
                             float* __restrict__ Gin_all) {
    const int dir = blockIdx.y;
    const _Float16* Wih = Wih_all + (long)dir * H4 * EP;
    const float*   bsum = bsum_all + dir * H4;
    float*         Gin  = Gin_all + (long)dir * TT * BB * H4;

    const int wv   = threadIdx.x >> 5;
    const int lane = threadIdx.x & 31;
    const int group = blockIdx.x * 8 + wv;   // 4096 groups: 256 mg x 16 ng
    const int m0 = (group >> 4) << 5;        // 2 M-tiles
    const int n0 = (group & 15) << 6;        // 4 N-tiles

    const int arow  = lane & 15;
    const int khalf = (lane >> 4) << 4;

    v8f acc[2][4];
#pragma unroll
    for (int mi = 0; mi < 2; ++mi)
#pragma unroll
        for (int nj = 0; nj < 4; ++nj) acc[mi][nj] = (v8f){};

#pragma unroll
    for (int k = 0; k < EP; k += 32) {
        v16h a0 = *(const v16h*)(X + (long)(m0 + arow) * EP + k + khalf);
        v16h a1 = *(const v16h*)(X + (long)(m0 + 16 + arow) * EP + k + khalf);
#pragma unroll
        for (int nj = 0; nj < 4; ++nj) {
            v16h b = *(const v16h*)(Wih + (long)(n0 + nj * 16 + arow) * EP + k + khalf);
            acc[0][nj] = WMMA_F16(a0, b, acc[0][nj]);
            acc[1][nj] = WMMA_F16(a1, b, acc[1][nj]);
        }
    }

    const int mrow = (lane >> 4) << 3;
#pragma unroll
    for (int mi = 0; mi < 2; ++mi) {
        const int mb = m0 + mi * 16 + mrow;
#pragma unroll
        for (int nj = 0; nj < 4; ++nj) {
            const int n = n0 + nj * 16 + (lane & 15);
#pragma unroll
            for (int v = 0; v < 8; ++v) {
                int r = mb + v;              // flat row b*T + t
                int t = r & (TT - 1);
                int b = r >> 7;
                Gin[((long)(t * BB + b)) * H4 + n] = acc[mi][nj][v] + bsum[n];
            }
        }
    }
}

// ---------------------------------------------------------------------------
// Serial LSTM recurrence: one workgroup per direction, 32 waves, 288KB dyn LDS.
// Each wave owns 1 M-tile x 8 N-tiles; the shared A (h-state) LDS fragment is
// loaded once per k-step and reused across all 8 WMMAs.
// ---------------------------------------------------------------------------
__global__ void k_lstm_recurrent(const float* __restrict__ Gin_all,
                                 const _Float16* __restrict__ Whh_all,
                                 _Float16* __restrict__ hs_all) {
    const int dir = blockIdx.x;
    const float*    Gin = Gin_all + (long)dir * TT * BB * H4;
    const _Float16* Whh = Whh_all + (long)dir * H4 * HH;
    _Float16*       hs  = hs_all  + (long)dir * BB * TT * HH;

    extern __shared__ char smem[];
    float*    g_s = (float*)smem;                          // 64*1024 f32 = 256KB
    _Float16* h_s = (_Float16*)(smem + (long)BB * H4 * 4); // 64*256 f16 = 32KB

    for (int i = threadIdx.x; i < BB * HH; i += blockDim.x) h_s[i] = (_Float16)0.0f;
    float c[16];
#pragma unroll
    for (int k = 0; k < 16; ++k) c[k] = 0.0f;

    const int wv    = threadIdx.x >> 5;
    const int lane  = threadIdx.x & 31;
    const int arow  = lane & 15;
    const int khalf = (lane >> 4) << 4;
    const int mt    = wv >> 3;                 // 4 M-tiles across 32 waves
    const int ntb   = (wv & 7) << 3;           // first of 8 N-tiles
    const int m0    = mt << 4;
    const int mb    = m0 + ((lane >> 4) << 3);
    const int nl    = lane & 15;
    const int j     = threadIdx.x & 255;
    const int bb0   = (threadIdx.x >> 8) << 4;
    __syncthreads();

    for (int s = 0; s < TT; ++s) {
        const int t = dir ? (TT - 1 - s) : s;
        // ---- g = Gin[t] + h @ Whh^T : 8 N-tiles per wave, A reused ----
        v8f acc[8];
#pragma unroll
        for (int i = 0; i < 8; ++i) {
            const int n = ((ntb + i) << 4) + nl;
            const float* gp = Gin + ((long)t * BB + mb) * H4 + n;
#pragma unroll
            for (int v = 0; v < 8; ++v) acc[i][v] = gp[(long)v * H4];
        }
#pragma unroll
        for (int k = 0; k < 8; ++k) {
            v16h a = *(const v16h*)(h_s + (m0 + arow) * HH + k * 32 + khalf);
#pragma unroll
            for (int i = 0; i < 8; ++i) {
                v16h b = *(const v16h*)(Whh + (long)(((ntb + i) << 4) + arow) * HH + k * 32 + khalf);
                acc[i] = WMMA_F16(a, b, acc[i]);
            }
        }
#pragma unroll
        for (int i = 0; i < 8; ++i) {
            const int n = ((ntb + i) << 4) + nl;
            float* gs = g_s + mb * H4 + n;
#pragma unroll
            for (int v = 0; v < 8; ++v) gs[v * H4] = acc[i][v];
        }
        __syncthreads();
        // ---- gates: i,f,g,o (PyTorch order); c in regs, h in LDS + global ----
#pragma unroll
        for (int k = 0; k < 16; ++k) {
            int b = bb0 + k;
            float ig = g_s[b * H4 + j];
            float fg = g_s[b * H4 + HH + j];
            float gg = g_s[b * H4 + 2 * HH + j];
            float og = g_s[b * H4 + 3 * HH + j];
            c[k] = sigf(fg) * c[k] + sigf(ig) * tanhf(gg);
            float h = sigf(og) * tanhf(c[k]);
            h_s[b * HH + j] = (_Float16)h;
            hs[((long)b * TT + t) * HH + j] = (_Float16)h;
        }
        __syncthreads();
    }
}

// ---------------------------------------------------------------------------
__global__ void k_masked_sum(const _Float16* __restrict__ hs_all,
                             const int* __restrict__ text_len,
                             _Float16* __restrict__ s16,
                             float* __restrict__ lenf) {
    int idx = blockIdx.x * blockDim.x + threadIdx.x;
    if (idx >= BB * GG) return;
    int b = idx >> 9;
    int g = idx & 511;
    int len = text_len[b];
    const _Float16* src = (g < HH)
        ? hs_all + (long)b * TT * HH + g
        : hs_all + (long)BB * TT * HH + (long)b * TT * HH + (g - HH);
    float acc = 0.0f;
    for (int t = 0; t < len; ++t) acc += (float)src[(long)t * HH];
    s16[idx] = (_Float16)acc;
    if (g == 0) lenf[b] = (float)len;
}

// ---------------------------------------------------------------------------
// Small WMMA GEMM head: out = relu?(A(MxK) @ W(NxK)^T + rowScale[m]*bias[n])
// ---------------------------------------------------------------------------
__global__ void k_head_gemm(const _Float16* __restrict__ A,
                            const _Float16* __restrict__ W,
                            const float* __restrict__ bias,
                            const float* __restrict__ rowScale,
                            _Float16* __restrict__ out,
                            int M, int N, int K, int relu) {
    const int wv   = threadIdx.x >> 5;
    const int lane = threadIdx.x & 31;
    const int nTiles = (M >> 4) * (N >> 4);
    const int tile = blockIdx.x * (blockDim.x >> 5) + wv;
    if (tile >= nTiles) return;
    const int ntN = N >> 4;
    const int m0 = (tile / ntN) << 4;
    const int n0 = (tile % ntN) << 4;
    const int arow  = lane & 15;
    const int khalf = (lane >> 4) << 4;

    v8f acc = {};
    for (int k = 0; k < K; k += 32) {
        v16h a = *(const v16h*)(A + (long)(m0 + arow) * K + k + khalf);
        v16h b = *(const v16h*)(W + (long)(n0 + arow) * K + k + khalf);
        acc = WMMA_F16(a, b, acc);
    }
    const int n  = n0 + (lane & 15);
    const int mb = m0 + ((lane >> 4) << 3);
#pragma unroll
    for (int v = 0; v < 8; ++v) {
        float rs  = rowScale ? rowScale[mb + v] : 1.0f;
        float val = acc[v] + rs * bias[n];
        if (relu) val = fmaxf(val, 0.0f);
        out[(long)(mb + v) * N + n] = (_Float16)val;
    }
}

__global__ void k_final_fc(const _Float16* __restrict__ x,
                           const float* __restrict__ w,
                           const float* __restrict__ bias,
                           float* __restrict__ out) {
    int idx = threadIdx.x;
    if (idx >= BB * NCC) return;
    int b = idx >> 1, c = idx & 1;
    float acc = bias[c];
    for (int jj = 0; jj < FHH; ++jj)
        acc += (float)x[(long)b * FHH + jj] * w[(long)c * FHH + jj];
    out[idx] = acc;
}

// ---------------------------------------------------------------------------

extern "C" void kernel_launch(void* const* d_in, const int* in_sizes, int n_in,
                              void* d_out, int out_size, void* d_ws, size_t ws_size,
                              hipStream_t stream) {
    const int*   sentence = (const int*)d_in[0];
    const int*   text_len = (const int*)d_in[4];
    const float* glove    = (const float*)d_in[10];
    const float* w_ih_f = (const float*)d_in[11];
    const float* w_hh_f = (const float*)d_in[12];
    const float* b_ih_f = (const float*)d_in[13];
    const float* b_hh_f = (const float*)d_in[14];
    const float* w_ih_b = (const float*)d_in[15];
    const float* w_hh_b = (const float*)d_in[16];
    const float* b_ih_b = (const float*)d_in[17];
    const float* b_hh_b = (const float*)d_in[18];
    const float* gat_w  = (const float*)d_in[19];
    const float* gat_b  = (const float*)d_in[20];
    const float* fc1_w  = (const float*)d_in[21];
    const float* fc1_b  = (const float*)d_in[22];
    const float* fc2_w  = (const float*)d_in[23];
    const float* fc2_b  = (const float*)d_in[24];
    const float* fcf_w  = (const float*)d_in[25];
    const float* fcf_b  = (const float*)d_in[26];

    char* ws = (char*)d_ws;
    size_t off = 0;
    auto take = [&](size_t bytes) -> char* {
        off = (off + 255) & ~(size_t)255;
        char* p = ws + off;
        off += bytes;
        return p;
    };
    _Float16* Xf16   = (_Float16*)take((size_t)BT * EP * 2);
    _Float16* Wih16  = (_Float16*)take((size_t)2 * H4 * EP * 2);
    _Float16* Whh16  = (_Float16*)take((size_t)2 * H4 * HH * 2);
    float*    bsum   = (float*)   take((size_t)2 * H4 * 4);
    float*    Gin    = (float*)   take((size_t)2 * TT * BB * H4 * 4);
    _Float16* hs     = (_Float16*)take((size_t)2 * BB * TT * HH * 2);
    _Float16* s16    = (_Float16*)take((size_t)BB * GG * 2);
    float*    lenf   = (float*)   take((size_t)BB * 4);
    _Float16* gw16   = (_Float16*)take((size_t)GG * GG * 2);
    _Float16* f1w16  = (_Float16*)take((size_t)FHH * GG * 2);
    _Float16* f2w16  = (_Float16*)take((size_t)FHH * FHH * 2);
    _Float16* x1     = (_Float16*)take((size_t)BB * GG * 2);
    _Float16* x2     = (_Float16*)take((size_t)BB * FHH * 2);
    _Float16* x3     = (_Float16*)take((size_t)BB * FHH * 2);

    {
        long n = (long)BT * EP;
        k_embed_f16<<<(int)((n + 255) / 256), 256, 0, stream>>>(sentence, glove, Xf16);
    }
    auto cvt = [&](const float* src, _Float16* dst, int rows, int sc, int dc) {
        long n = (long)rows * dc;
        k_cvt_pad_f16<<<(int)((n + 255) / 256), 256, 0, stream>>>(src, dst, rows, sc, dc);
    };
    cvt(w_ih_f, Wih16,               H4, EE, EP);
    cvt(w_ih_b, Wih16 + (long)H4*EP, H4, EE, EP);
    cvt(w_hh_f, Whh16,               H4, HH, HH);
    cvt(w_hh_b, Whh16 + (long)H4*HH, H4, HH, HH);
    cvt(gat_w,  gw16,  GG,  GG,  GG);
    cvt(fc1_w,  f1w16, FHH, GG,  GG);
    cvt(fc2_w,  f2w16, FHH, FHH, FHH);
    k_bias_sum<<<8, 256, 0, stream>>>(b_ih_f, b_hh_f, b_ih_b, b_hh_b, bsum);

    // 4096 wave-groups per dir, 8 waves per block -> 512 blocks x 2 dirs
    k_input_gemm<<<dim3(512, 2), 256, 0, stream>>>(Xf16, Wih16, bsum, Gin);

    k_lstm_recurrent<<<2, 1024, (size_t)BB * H4 * 4 + (size_t)BB * HH * 2, stream>>>(Gin, Whh16, hs);

    k_masked_sum<<<(BB * GG + 255) / 256, 256, 0, stream>>>(hs, text_len, s16, lenf);

    k_head_gemm<<<16, 256, 0, stream>>>(s16, gw16, gat_b, lenf,    x1, BB, GG,  GG,  1);
    k_head_gemm<<<8,  256, 0, stream>>>(x1,  f1w16, fc1_b, nullptr, x2, BB, FHH, GG,  1);
    k_head_gemm<<<8,  256, 0, stream>>>(x2,  f2w16, fc2_b, nullptr, x3, BB, FHH, FHH, 1);
    k_final_fc<<<1, 128, 0, stream>>>(x3, fcf_w, fcf_b, (float*)d_out);
}
